// CorrectionNet_77438260346965
// MI455X (gfx1250) — matrix operand order repaired and checked by reading, not compiled, not run on hardware
//
#include <hip/hip_runtime.h>

typedef __attribute__((ext_vector_type(16))) _Float16 v16h;
typedef __attribute__((ext_vector_type(8)))  float    v8f;

#define HDIM 16
#define TPW  4   // tiles per wave (grid-stride amortization)

// per-wave LDS ordering: CDNA5 split dependency counter for DS ops
#define LDS_SYNC() asm volatile("s_wait_dscnt 0" ::: "memory")

// ---------------------------------------------------------------------------
// WMMA fragment helpers (wave32, 16x16 tiles, f16 A/B, f32 C/D)
// A 16x32 f16: lane L holds row (L&15), K = (L>>4)*8 + {0..7, 16..23}
// B 32x16 f16: lane L holds col (L&15), K = (L>>4)*8 + {0..7, 16..23}
// C/D 16x16 f32: lane L holds col (L&15), rows (L>>4)*8 + r
// ---------------------------------------------------------------------------

__device__ __forceinline__ v16h load_b_frag(const float* __restrict__ W,
                                            int krows, int kofs, int lane) {
  int col = lane & 15;
  int kb  = kofs + ((lane >> 4) << 3);
  v16h b;
#pragma unroll
  for (int e = 0; e < 16; ++e) {
    int k   = kb + (e < 8 ? e : e + 8);
    float w = (k < krows) ? W[k * HDIM + col] : 0.0f;
    b[e] = (_Float16)w;
  }
  return b;
}

__device__ __forceinline__ v16h load_a_frag(const _Float16* __restrict__ p,
                                            int stride, int kofs, int lane) {
  int row = lane & 15;
  int kb  = kofs + ((lane >> 4) << 3);
  const _Float16* q = p + row * stride + kb;
  v16h a;
#pragma unroll
  for (int e = 0; e < 8; ++e) { a[e] = q[e]; a[e + 8] = q[e + 16]; }
  return a;
}

__device__ __forceinline__ v8f bias_frag(const float* __restrict__ b, int lane) {
  float v = b[lane & 15];
  v8f c;
#pragma unroll
  for (int r = 0; r < 8; ++r) c[r] = v;
  return c;
}

// Stage one contiguous 16x16 f32 tile (256 floats) into LDS rows as f16 at
// column offset cofs. Fully coalesced: each lane does 2 global b128 loads and
// 2 packed 8-byte LDS stores.
__device__ __forceinline__ void stage_tile16(const float* __restrict__ src,
                                             _Float16* __restrict__ dst,
                                             int rowstride, int cofs, int lane) {
#pragma unroll
  for (int i = 0; i < 2; ++i) {
    int v   = lane + i * 32;        // float4 chunk 0..63
    int row = v >> 2;
    int c4  = (v & 3) << 2;
    float4 f = ((const float4*)src)[v];
    _Float16* q = dst + row * rowstride + cofs + c4;
    q[0] = (_Float16)f.x; q[1] = (_Float16)f.y;
    q[2] = (_Float16)f.z; q[3] = (_Float16)f.w;
  }
}

// Stage one random-indexed 16-float row into LDS row m at column offset cofs:
// 4 global b128 loads + packed LDS stores.
__device__ __forceinline__ void stage_row16(const float* __restrict__ src,
                                            _Float16* __restrict__ dstrow,
                                            int cofs) {
  const float4* p = (const float4*)src;
#pragma unroll
  for (int j = 0; j < 4; ++j) {
    float4 f = p[j];
    _Float16* q = dstrow + cofs + j * 4;
    q[0] = (_Float16)f.x; q[1] = (_Float16)f.y;
    q[2] = (_Float16)f.z; q[3] = (_Float16)f.w;
  }
}

// Zero a 16-row x 16-half pad region once per wave (vectorized: 16B per lane).
__device__ __forceinline__ void zero_pad16(_Float16* dst, int rowstride,
                                           int cofs, int lane) {
  _Float16* q = dst + (lane >> 1) * rowstride + cofs + (lane & 1) * 8;
#pragma unroll
  for (int e = 0; e < 8; ++e) q[e] = (_Float16)0.0f;
}

// ---------------------------------------------------------------------------
// norm = max |edges_init|
// ---------------------------------------------------------------------------
__global__ void absmax_kernel(const float* __restrict__ x, unsigned* normbits, int n) {
  int i = blockIdx.x * blockDim.x + threadIdx.x;
  float v = (i < n) ? fabsf(x[i]) : 0.0f;
#pragma unroll
  for (int off = 16; off > 0; off >>= 1)
    v = fmaxf(v, __shfl_xor(v, off, 32));
  if ((threadIdx.x & 31) == 0)
    atomicMax(normbits, __float_as_uint(v));
}

// ---------------------------------------------------------------------------
// Encoder: h = relu(x*W1 + b1) @ W2 + b2, x is [rows,1]
// layer 1 is K=1 (VALU outer product), layer 2 is one WMMA (K=16 padded to 32)
// ---------------------------------------------------------------------------
__global__ void encode_kernel(const float* __restrict__ x,
                              float* __restrict__ hout,
                              const float* __restrict__ W1,  // [1,16]
                              const float* __restrict__ b1,
                              const float* __restrict__ W2,  // [16,16]
                              const float* __restrict__ b2,
                              const float* __restrict__ normptr,  // null => scale 1
                              int ntiles) {
  __shared__ _Float16 sact[4][16][32];
  int wave = threadIdx.x >> 5;
  int lane = threadIdx.x & 31;
  int gw   = blockIdx.x * 4 + wave;
  int nw   = gridDim.x * 4;

  float scale = normptr ? (1.0f / *normptr) : 1.0f;
  v16h wb2 = load_b_frag(W2, 16, 0, lane);
  zero_pad16(&sact[wave][0][0], 32, 16, lane);

  int col = lane & 15, rbase = (lane >> 4) << 3;
  for (int tile = gw; tile < ntiles; tile += nw) {
    int tb = tile * 16;
    if (lane < 16) {
      float xv = x[tb + lane] * scale;
#pragma unroll
      for (int h = 0; h < HDIM; ++h)
        sact[wave][lane][h] = (_Float16)fmaxf(fmaf(xv, W1[h], b1[h]), 0.0f);
    }
    LDS_SYNC();
    v16h a = load_a_frag(&sact[wave][0][0], 32, 0, lane);
    v8f acc = bias_frag(b2, lane);
    acc = __builtin_amdgcn_wmma_f32_16x16x32_f16(false, a, false, wb2,
                                                 (short)0, acc, false, false);
#pragma unroll
    for (int r = 0; r < 8; ++r)
      hout[(tb + rbase + r) * HDIM + col] = acc[r];
    LDS_SYNC();   // finish reads before next tile overwrites sact
  }
}

// ---------------------------------------------------------------------------
// Edge MLP round: h_e = MLP2([h_n[send], h_n[recv], h_e]) + atomic scatter to agg
// ---------------------------------------------------------------------------
__global__ void edge_mlp_kernel(const float* __restrict__ hn,
                                float* __restrict__ he,
                                float* __restrict__ agg,
                                const int* __restrict__ senders,
                                const int* __restrict__ receivers,
                                const float* __restrict__ W1,  // [48,16]
                                const float* __restrict__ b1,
                                const float* __restrict__ W2,  // [16,16]
                                const float* __restrict__ b2,
                                int ntiles) {
  __shared__ _Float16 smsg[4][16][64];
  __shared__ _Float16 sact[4][16][32];
  __shared__ int      sridx[4][16];
  int wave = threadIdx.x >> 5;
  int lane = threadIdx.x & 31;
  int gw   = blockIdx.x * 4 + wave;
  int nw   = gridDim.x * 4;

  v16h wb1a = load_b_frag(W1, 48, 0,  lane);  // K rows 0..31
  v16h wb1b = load_b_frag(W1, 48, 32, lane);  // K rows 32..47 (+pad)
  v16h wb2  = load_b_frag(W2, 16, 0,  lane);  // K rows 0..15 (+pad)
  zero_pad16(&smsg[wave][0][0], 64, 48, lane);  // msg K-pad 48..63
  zero_pad16(&sact[wave][0][0], 32, 16, lane);  // act K-pad 16..31

  int m = lane & 15;
  int col = lane & 15, rbase = (lane >> 4) << 3;
  bool isRecv = lane >= 16;

  for (int tile = gw; tile < ntiles; tile += nw) {
    int tb = tile * 16;
    // gather: lanes 0-15 stage sender rows, lanes 16-31 receiver rows;
    // all 32 lanes then stage the contiguous h_e tile coalesced.
    int idx = isRecv ? receivers[tb + m] : senders[tb + m];
    if (isRecv) sridx[wave][m] = idx;
    stage_row16(hn + (size_t)idx * HDIM, &smsg[wave][m][0], isRecv ? 16 : 0);
    stage_tile16(he + (size_t)tb * HDIM, &smsg[wave][0][0], 64, 32, lane);
    LDS_SYNC();

    v16h a0 = load_a_frag(&smsg[wave][0][0], 64, 0,  lane);
    v16h a1 = load_a_frag(&smsg[wave][0][0], 64, 32, lane);
    v8f acc = bias_frag(b1, lane);
    acc = __builtin_amdgcn_wmma_f32_16x16x32_f16(false, a0, false, wb1a,
                                                 (short)0, acc, false, false);
    acc = __builtin_amdgcn_wmma_f32_16x16x32_f16(false, a1, false, wb1b,
                                                 (short)0, acc, false, false);
#pragma unroll
    for (int r = 0; r < 8; ++r)
      sact[wave][rbase + r][col] = (_Float16)fmaxf(acc[r], 0.0f);
    LDS_SYNC();

    v16h a2 = load_a_frag(&sact[wave][0][0], 32, 0, lane);
    v8f out = bias_frag(b2, lane);
    out = __builtin_amdgcn_wmma_f32_16x16x32_f16(false, a2, false, wb2,
                                                 (short)0, out, false, false);
#pragma unroll
    for (int r = 0; r < 8; ++r) {
      int row = rbase + r;
      he[(tb + row) * HDIM + col] = out[r];                    // in-place update
      atomicAdd(&agg[(size_t)sridx[wave][row] * HDIM + col], out[r]);  // segment_sum
    }
    LDS_SYNC();   // finish fragment reads before next tile's staging
  }
}

// ---------------------------------------------------------------------------
// Node MLP round: h_n = MLP2([h_n, agg])  (in place)
// ---------------------------------------------------------------------------
__global__ void node_mlp_kernel(float* __restrict__ hn,
                                const float* __restrict__ agg,
                                const float* __restrict__ W1,  // [32,16]
                                const float* __restrict__ b1,
                                const float* __restrict__ W2,  // [16,16]
                                const float* __restrict__ b2,
                                int ntiles) {
  __shared__ _Float16 smsg[4][16][32];
  __shared__ _Float16 sact[4][16][32];
  int wave = threadIdx.x >> 5;
  int lane = threadIdx.x & 31;
  int gw   = blockIdx.x * 4 + wave;
  int nw   = gridDim.x * 4;

  v16h wb1 = load_b_frag(W1, 32, 0, lane);
  v16h wb2 = load_b_frag(W2, 16, 0, lane);
  zero_pad16(&sact[wave][0][0], 32, 16, lane);

  int col = lane & 15, rbase = (lane >> 4) << 3;
  for (int tile = gw; tile < ntiles; tile += nw) {
    int tb = tile * 16;
    stage_tile16(hn  + (size_t)tb * HDIM, &smsg[wave][0][0], 32, 0,  lane);
    stage_tile16(agg + (size_t)tb * HDIM, &smsg[wave][0][0], 32, 16, lane);
    LDS_SYNC();

    v16h a0 = load_a_frag(&smsg[wave][0][0], 32, 0, lane);
    v8f acc = bias_frag(b1, lane);
    acc = __builtin_amdgcn_wmma_f32_16x16x32_f16(false, a0, false, wb1,
                                                 (short)0, acc, false, false);
#pragma unroll
    for (int r = 0; r < 8; ++r)
      sact[wave][rbase + r][col] = (_Float16)fmaxf(acc[r], 0.0f);
    LDS_SYNC();

    v16h a1 = load_a_frag(&sact[wave][0][0], 32, 0, lane);
    v8f out = bias_frag(b2, lane);
    out = __builtin_amdgcn_wmma_f32_16x16x32_f16(false, a1, false, wb2,
                                                 (short)0, out, false, false);
#pragma unroll
    for (int r = 0; r < 8; ++r)
      hn[(tb + rbase + r) * HDIM + col] = out[r];
    LDS_SYNC();
  }
}

// ---------------------------------------------------------------------------
// Bidirectional edge averaging: pairs partition the edge set -> race free
// ---------------------------------------------------------------------------
__global__ void biavg_kernel(float* __restrict__ he,
                             const int* __restrict__ bi, int npairs) {
  int t = blockIdx.x * blockDim.x + threadIdx.x;
  if (t >= npairs) return;
  int i0 = bi[2 * t], i1 = bi[2 * t + 1];
  float4* p0 = (float4*)(he + (size_t)i0 * HDIM);
  float4* p1 = (float4*)(he + (size_t)i1 * HDIM);
#pragma unroll
  for (int j = 0; j < 4; ++j) {
    float4 a = p0[j], b = p1[j];
    float4 v;
    v.x = 0.5f * (a.x + b.x); v.y = 0.5f * (a.y + b.y);
    v.z = 0.5f * (a.z + b.z); v.w = 0.5f * (a.w + b.w);
    p0[j] = v; p1[j] = v;
  }
}

// ---------------------------------------------------------------------------
// Decoder: out = mask(senders>=receivers) * (edges_init + alpha*norm*MLP2(h_e))
// layer1 = WMMA, layer2 (16 -> 1) = VALU dot, fused residual + tril mask
// ---------------------------------------------------------------------------
__global__ void decode_kernel(const float* __restrict__ he,
                              const float* __restrict__ edges_init,
                              const int* __restrict__ senders,
                              const int* __restrict__ receivers,
                              const float* __restrict__ W1,  // [16,16]
                              const float* __restrict__ b1,
                              const float* __restrict__ W2,  // [16,1]
                              const float* __restrict__ b2,  // [1]
                              const float* __restrict__ normptr,
                              const float* __restrict__ alphaptr,
                              float* __restrict__ out, int ntiles) {
  __shared__ _Float16 smsg[4][16][32];
  __shared__ float    sres[4][16][16];
  int wave = threadIdx.x >> 5;
  int lane = threadIdx.x & 31;
  int gw   = blockIdx.x * 4 + wave;
  int nw   = gridDim.x * 4;

  v16h wb1 = load_b_frag(W1, 16, 0, lane);
  zero_pad16(&smsg[wave][0][0], 32, 16, lane);
  float nrm = *normptr, alf = *alphaptr;

  int col = lane & 15, rbase = (lane >> 4) << 3;
  for (int tile = gw; tile < ntiles; tile += nw) {
    int tb = tile * 16;
    stage_tile16(he + (size_t)tb * HDIM, &smsg[wave][0][0], 32, 0, lane);
    LDS_SYNC();

    v16h a = load_a_frag(&smsg[wave][0][0], 32, 0, lane);
    v8f acc = bias_frag(b1, lane);
    acc = __builtin_amdgcn_wmma_f32_16x16x32_f16(false, a, false, wb1,
                                                 (short)0, acc, false, false);
#pragma unroll
    for (int r = 0; r < 8; ++r)
      sres[wave][rbase + r][col] = fmaxf(acc[r], 0.0f);
    LDS_SYNC();

    if (lane < 16) {
      float s = b2[0];
#pragma unroll
      for (int h = 0; h < HDIM; ++h) s = fmaf(sres[wave][lane][h], W2[h], s);
      int e = tb + lane;
      float o = 0.0f;
      if (senders[e] >= receivers[e])
        o = edges_init[e] + alf * (s * nrm);
      out[e] = o;
    }
    LDS_SYNC();
  }
}

// ---------------------------------------------------------------------------
extern "C" void kernel_launch(void* const* d_in, const int* in_sizes, int n_in,
                              void* d_out, int out_size, void* d_ws, size_t ws_size,
                              hipStream_t stream) {
  const float* nodes      = (const float*)d_in[0];
  const float* edges_init = (const float*)d_in[1];
  const int*   receivers  = (const int*)d_in[2];
  const int*   senders    = (const int*)d_in[3];
  const int*   bi_idx     = (const int*)d_in[4];
  const float* node_W1 = (const float*)d_in[9],  *node_b1 = (const float*)d_in[10];
  const float* node_W2 = (const float*)d_in[11], *node_b2 = (const float*)d_in[12];
  const float* edge_W1 = (const float*)d_in[13], *edge_b1 = (const float*)d_in[14];
  const float* edge_W2 = (const float*)d_in[15], *edge_b2 = (const float*)d_in[16];
  const float* mpe_W1  = (const float*)d_in[17], *mpe_b1  = (const float*)d_in[18];
  const float* mpe_W2  = (const float*)d_in[19], *mpe_b2  = (const float*)d_in[20];
  const float* mpn_W1  = (const float*)d_in[21], *mpn_b1  = (const float*)d_in[22];
  const float* mpn_W2  = (const float*)d_in[23], *mpn_b2  = (const float*)d_in[24];
  const float* dec_W1  = (const float*)d_in[25], *dec_b1  = (const float*)d_in[26];
  const float* dec_W2  = (const float*)d_in[27], *dec_b2  = (const float*)d_in[28];
  const float* alpha   = (const float*)d_in[29];

  const int N = in_sizes[0];      // 262144
  const int E = in_sizes[1];      // 1310720
  const int NT_N = N / 16;
  const int NT_E = E / 16;
  const int WPT  = 4 * TPW;       // waves-per-block * tiles-per-wave
  const int GB_N = (NT_N + WPT - 1) / WPT;
  const int GB_E = (NT_E + WPT - 1) / WPT;

  // workspace layout: [norm u32 (256B pad)] [h_n N*16] [agg N*16] [h_e E*16]
  unsigned* normbits = (unsigned*)d_ws;
  float* hn  = (float*)((char*)d_ws + 256);
  float* agg = hn + (size_t)N * HDIM;
  float* he  = agg + (size_t)N * HDIM;

  hipMemsetAsync(normbits, 0, sizeof(unsigned), stream);
  absmax_kernel<<<(E + 255) / 256, 256, 0, stream>>>(edges_init, normbits, E);

  encode_kernel<<<GB_N, 128, 0, stream>>>(
      nodes, hn, node_W1, node_b1, node_W2, node_b2, nullptr, NT_N);
  encode_kernel<<<GB_E, 128, 0, stream>>>(
      edges_init, he, edge_W1, edge_b1, edge_W2, edge_b2,
      (const float*)normbits, NT_E);

  for (int t = 0; t < 5; ++t) {
    hipMemsetAsync(agg, 0, (size_t)N * HDIM * sizeof(float), stream);
    edge_mlp_kernel<<<GB_E, 128, 0, stream>>>(
        hn, he, agg, senders, receivers, mpe_W1, mpe_b1, mpe_W2, mpe_b2, NT_E);
    node_mlp_kernel<<<GB_N, 128, 0, stream>>>(
        hn, agg, mpn_W1, mpn_b1, mpn_W2, mpn_b2, NT_N);
  }

  biavg_kernel<<<(E / 2 + 255) / 256, 256, 0, stream>>>(he, bi_idx, E / 2);

  decode_kernel<<<GB_E, 128, 0, stream>>>(
      he, edges_init, senders, receivers, dec_W1, dec_b1, dec_W2, dec_b2,
      (const float*)normbits, alpha, (float*)d_out, NT_E);
}
